// ETC_22514218566140
// MI455X (gfx1250) — compile-verified, matrix operand order checked by reading, not run on hardware
//
#include <hip/hip_runtime.h>

// Problem constants (from the reference)
#define T_TIMES   200
#define K_TOPICS  512
#define D_DIM     128
#define INV_TEMP  2.0f      // 1 / TEMPERATURE
#define EPS_F     1e-12f
#define W_NEG     1.0f
#define W_POS     1.0f

typedef __attribute__((ext_vector_type(16))) _Float16 v16h;
typedef __attribute__((ext_vector_type(8)))  _Float16 v8h;
typedef __attribute__((ext_vector_type(8)))  float    v8f;

// Load one 16x32 f16 WMMA A/B fragment from a row-major [16 x 128] LDS strip.
// CDNA5 16-bit A layout: lane L in [0,16) holds row M=L with
// K = {b..b+7, b+16..b+23}, b = 0 for lanes 0-15, b = 8 for lanes 16-31.
// B (32x16 = contrast^T) mirrors it with N <-> row of the row-major matrix.
static __device__ inline v16h load_frag(const _Float16* sm, int row, int kstep, int lane) {
    const int base = row * D_DIM + ((lane & 16) ? 8 : 0) + kstep * 32;
    v8h lo = *reinterpret_cast<const v8h*>(sm + base);        // ds_load_b128
    v8h hi = *reinterpret_cast<const v8h*>(sm + base + 16);   // ds_load_b128
    v16h out;
#pragma unroll
    for (int i = 0; i < 8; ++i) { out[i] = lo[i]; out[i + 8] = hi[i]; }
    return out;
}

// Fetch all 4 K-step fragments of one 16x16 output tile's B operand.
static __device__ inline void load_tile(v16h (&b)[4], const _Float16* smB, int jb, int lane) {
    const int rowB = jb * 16 + (lane & 15);
#pragma unroll
    for (int ks = 0; ks < 4; ++ks) b[ks] = load_frag(smB, rowB, ks, lane);
}

// 4x WMMA (full D=128 contraction) + per-tile epilogue.
template <bool IS_POS>
static __device__ inline void process_tile(const v16h (&aF)[4], const v16h (&bF)[4],
                                           int kb, int jb, int lane,
                                           float (&acc)[8], float& diagSum) {
    v8f c = {};
#pragma unroll
    for (int ks = 0; ks < 4; ++ks)
        c = __builtin_amdgcn_wmma_f32_16x16x32_f16(
                /*neg_a=*/false, aF[ks], /*neg_b=*/false, bF[ks],
                /*c_mod=*/(short)0, c, /*reuse_a=*/false, /*reuse_b=*/false);
    // C/D layout: VGPR r -> row M = r (+8 for lanes>=16); lane&15 -> col N
#pragma unroll
    for (int r = 0; r < 8; ++r) {
        const float logit  = c[r] * INV_TEMP;
        const bool  isDiag = (jb == kb) &&
                             ((lane & 15) == r + ((lane & 16) ? 8 : 0));
        if (IS_POS) {
            acc[r] = fmaxf(acc[r], logit);
            if (isDiag) diagSum += logit;
        } else {
            acc[r] += isDiag ? 0.0f : __expf(logit - INV_TEMP);  // lmax == 1/TEMP exactly
        }
    }
}

// L2-normalize rows of 128 fp32 into f16 LDS. One wave per row, float4 per
// lane, full-wave32 shfl_xor reduction for the squared norm.
static __device__ inline void normalize_rows(const float* __restrict__ src,
                                             _Float16* dst,
                                             int wave, int lane) {
    for (int row = wave; row < K_TOPICS; row += 16) {
        const float4 x4 = reinterpret_cast<const float4*>(src + (size_t)row * D_DIM)[lane];
        float ss = x4.x * x4.x + x4.y * x4.y + x4.z * x4.z + x4.w * x4.w;
        ss += __shfl_xor(ss, 1);
        ss += __shfl_xor(ss, 2);
        ss += __shfl_xor(ss, 4);
        ss += __shfl_xor(ss, 8);
        ss += __shfl_xor(ss, 16);
        const float inv = 1.0f / fmaxf(sqrtf(ss), EPS_F);
        _Float16* d = dst + row * D_DIM + lane * 4;
        d[0] = (_Float16)(x4.x * inv);
        d[1] = (_Float16)(x4.y * inv);
        d[2] = (_Float16)(x4.z * inv);
        d[3] = (_Float16)(x4.w * inv);
    }
}

// IS_POS == false: block t computes Gram(norm[t], norm[t]) and
//   partial[t] = sum_k log( sum_{j!=k} exp(2*g_kj - 2) + eps )
// IS_POS == true: block t computes Gram(norm[t+1], norm[t]) and
//   partial[t] = sum_k ( pmax_k - 2*g'_kk )
template <bool IS_POS>
__global__ __launch_bounds__(512) void gram_loss_kernel(const float* __restrict__ te,
                                                        float* __restrict__ partial) {
    extern __shared__ __align__(16) char smem[];
    _Float16* smA = reinterpret_cast<_Float16*>(smem);
    _Float16* smB = IS_POS ? (smA + K_TOPICS * D_DIM) : smA;
    __shared__ float red[16];

    const int t    = blockIdx.x;
    const int lane = threadIdx.x & 31;
    const int wave = threadIdx.x >> 5;

    // ---- Phase 1: normalize into LDS (f16) ----
    const float* srcA = te + (size_t)(IS_POS ? (t + 1) : t) * K_TOPICS * D_DIM;
    normalize_rows(srcA, smA, wave, lane);
    if (IS_POS) {
        const float* srcB = te + (size_t)t * K_TOPICS * D_DIM;
        normalize_rows(srcB, smB, wave, lane);
    }
    __syncthreads();

    // ---- Phase 2: tiled Gram via v_wmma_f32_16x16x32_f16, B double-buffered ----
    float wavePartial = 0.0f;   // lanes 0/16 carry per-row log / pmax terms
    float diagSum     = 0.0f;   // POS only: sum of diagonal logits (per lane)

    for (int kb = wave; kb < 32; kb += 16) {     // two 16-row strips per wave
        const int rowA = kb * 16 + (lane & 15);
        v16h aF[4];
#pragma unroll
        for (int ks = 0; ks < 4; ++ks) aF[ks] = load_frag(smA, rowA, ks, lane);

        float acc[8];
#pragma unroll
        for (int r = 0; r < 8; ++r) acc[r] = IS_POS ? -3.402823466e38f : 0.0f;

        // Software pipeline: fetch tile j+1's B fragments while tile j's
        // WMMAs run, so the backend can emit partial s_wait_dscnt.
        v16h bufA[4], bufB[4];
        load_tile(bufA, smB, 0, lane);
        for (int jb = 0; jb < 32; jb += 2) {
            load_tile(bufB, smB, jb + 1, lane);
            process_tile<IS_POS>(aF, bufA, kb, jb, lane, acc, diagSum);
            load_tile(bufA, smB, (jb + 2) & 31, lane);   // wrap: harmless reload at end
            process_tile<IS_POS>(aF, bufB, kb, jb + 1, lane, acc, diagSum);
        }

        // Row reduction across the 16 lanes of each half-wave
#pragma unroll
        for (int r = 0; r < 8; ++r) {
            float v = acc[r];
            if (IS_POS) {
                v = fmaxf(v, __shfl_xor(v, 1));
                v = fmaxf(v, __shfl_xor(v, 2));
                v = fmaxf(v, __shfl_xor(v, 4));
                v = fmaxf(v, __shfl_xor(v, 8));
                if ((lane & 15) == 0) wavePartial += v;                 // +pmax_row
            } else {
                v += __shfl_xor(v, 1);
                v += __shfl_xor(v, 2);
                v += __shfl_xor(v, 4);
                v += __shfl_xor(v, 8);
                if ((lane & 15) == 0) wavePartial += __logf(v + EPS_F); // log sum-exp term
            }
        }
    }

    // ---- Phase 3: deterministic block reduction ----
    float lanePart = wavePartial - (IS_POS ? diagSum : 0.0f);
    lanePart += __shfl_xor(lanePart, 1);
    lanePart += __shfl_xor(lanePart, 2);
    lanePart += __shfl_xor(lanePart, 4);
    lanePart += __shfl_xor(lanePart, 8);
    lanePart += __shfl_xor(lanePart, 16);
    if (lane == 0) red[wave] = lanePart;
    __syncthreads();
    if (threadIdx.x == 0) {
        float s = 0.0f;
#pragma unroll
        for (int w = 0; w < 16; ++w) s += red[w];   // fixed order -> deterministic
        partial[t] = s;
    }
}

__global__ void finalize_kernel(const float* __restrict__ negP,
                                const float* __restrict__ posP,
                                float* __restrict__ out) {
    if (threadIdx.x == 0 && blockIdx.x == 0) {
        float negT = 0.0f;
        for (int t = 0; t < T_TIMES; ++t) negT += negP[t];
        float posT = 0.0f;
        for (int t = 0; t < T_TIMES - 1; ++t) posT += posP[t];
        const float loss_neg = negT / (float)(K_TOPICS * (K_TOPICS - 1));
        const float loss_pos = posT / (float)K_TOPICS;
        out[0] = loss_neg * (W_NEG / (float)T_TIMES) +
                 loss_pos * (W_POS / (float)(T_TIMES - 1));
    }
}

extern "C" void kernel_launch(void* const* d_in, const int* in_sizes, int n_in,
                              void* d_out, int out_size, void* d_ws, size_t ws_size,
                              hipStream_t stream) {
    (void)in_sizes; (void)n_in; (void)out_size; (void)ws_size;
    const float* te = (const float*)d_in[0];
    float* wsF  = (float*)d_ws;
    float* negP = wsF;              // T entries
    float* posP = wsF + T_TIMES;    // T-1 entries

    const size_t smemNeg = (size_t)K_TOPICS * D_DIM * sizeof(_Float16);     // 128 KB
    const size_t smemPos = 2 * smemNeg;                                     // 256 KB

    gram_loss_kernel<false><<<T_TIMES,     512, smemNeg, stream>>>(te, negP);
    gram_loss_kernel<true ><<<T_TIMES - 1, 512, smemPos, stream>>>(te, posP);
    finalize_kernel<<<1, 32, 0, stream>>>(negP, posP, (float*)d_out);
}
